// ExpertLinear_2070174236928
// MI455X (gfx1250) — matrix operand order repaired and checked by reading, not compiled, not run on hardware
//
#include <hip/hip_runtime.h>
#include <hip/hip_bf16.h>

// MoE-weighted linear: y[b,o] = sum_e w[b,e] * (x[b,:] @ W[e] + bias[e,o])
// Strategy: y = sum_e (diag(w[:,e]) x) @ W[e]  +  w @ bias
// bf16 WMMA (v_wmma_f32_16x16x32_bf16), f32 accumulate; expert scale folded
// into the fp32->bf16 convert when staging A into LDS.
// Compute-bound (275 GFLOP); all operands fit in the 192MB L2 so re-reads are cheap.

typedef __bf16 bf16_t;
typedef __bf16 v16bf __attribute__((ext_vector_type(16)));
typedef __bf16 v8bf  __attribute__((ext_vector_type(8)));
typedef __bf16 v2bf  __attribute__((ext_vector_type(2)));
typedef float  v8f   __attribute__((ext_vector_type(8)));

#define EXPERTS 8
#define IN_DIM  1024
#define OUT_DIM 1024
#define BATCH   16384

#define BM 128
#define BN 128
#define BK 32
#define ASTR 40   // padded LDS row stride for A [BM][BK] (bf16 elems)
#define BSTR 40   // padded LDS col stride for Bt [BN][BK] (bf16 elems)

__global__ __launch_bounds__(256)
void moe_linear_bf16_wmma(const float* __restrict__ x,
                          const float* __restrict__ gate,   // [B, E]
                          const float* __restrict__ W,      // [E, K, N]
                          const float* __restrict__ bias,   // [E, 1, N]
                          float* __restrict__ out)          // [B, N]
{
    __shared__ __align__(16) bf16_t Ash[BM * ASTR];  // row-major [row][k], pre-scaled by gate
    __shared__ __align__(16) bf16_t Bsh[BN * BSTR];  // transposed  [n][k]

    const int tid  = threadIdx.x;
    const int lane = tid & 31;
    const int wid  = tid >> 5;      // 0..7
    const int wm   = wid & 3;       // 4 M-slices of 32 rows
    const int wn   = wid >> 2;      // 2 N-slices of 64 cols
    const int hlf  = lane >> 4;     // lane half (K-half select per WMMA layout)
    const int l16  = lane & 15;

    const int blockN = blockIdx.x * BN;
    const int blockM = blockIdx.y * BM;

    // ---- tile-fill work assignment (256 threads) ----
    const int aRow = tid >> 1;           // 0..127
    const int aSeg = (tid & 1) * 16;     // 0 or 16 (k-half of 32)
    const int bK   = (tid >> 4) * 2;     // 0,2,...,30 (k-pair)
    const int bN8  = (tid & 15) * 8;     // 0,8,...,120 (n-group)

    v8f acc[2][4];
    {
        v8f zero = {};
        #pragma unroll
        for (int i = 0; i < 2; ++i)
            #pragma unroll
            for (int j = 0; j < 4; ++j) acc[i][j] = zero;
    }

    for (int e = 0; e < EXPERTS; ++e) {
        const float* We = W + (size_t)e * IN_DIM * OUT_DIM;
        const float aScale = gate[(size_t)(blockM + aRow) * EXPERTS + e];

        for (int k0 = 0; k0 < IN_DIM; k0 += BK) {
            // ---------- stage A: 128x32 fp32 -> scaled bf16, row-major ----------
            {
                const float4* xp =
                    (const float4*)(x + (size_t)(blockM + aRow) * IN_DIM + k0 + aSeg);
                float4 f0 = xp[0], f1 = xp[1], f2 = xp[2], f3 = xp[3];
                float fs[16] = { f0.x, f0.y, f0.z, f0.w,  f1.x, f1.y, f1.z, f1.w,
                                 f2.x, f2.y, f2.z, f2.w,  f3.x, f3.y, f3.z, f3.w };
                v8bf lo, hi;
                #pragma unroll
                for (int j = 0; j < 8; ++j) {
                    lo[j] = (bf16_t)(fs[j]     * aScale);
                    hi[j] = (bf16_t)(fs[j + 8] * aScale);
                }
                *(v8bf*)&Ash[aRow * ASTR + aSeg]     = lo;
                *(v8bf*)&Ash[aRow * ASTR + aSeg + 8] = hi;
            }
            // ---------- stage B: 32x128 fp32 -> bf16, transposed to [n][k] ----------
            {
                const float* wr0 = We + (size_t)(k0 + bK)     * OUT_DIM + blockN + bN8;
                const float* wr1 = We + (size_t)(k0 + bK + 1) * OUT_DIM + blockN + bN8;
                float4 r00 = ((const float4*)wr0)[0], r01 = ((const float4*)wr0)[1];
                float4 r10 = ((const float4*)wr1)[0], r11 = ((const float4*)wr1)[1];
                float c0[8] = { r00.x, r00.y, r00.z, r00.w, r01.x, r01.y, r01.z, r01.w };
                float c1[8] = { r10.x, r10.y, r10.z, r10.w, r11.x, r11.y, r11.z, r11.w };
                #pragma unroll
                for (int j = 0; j < 8; ++j) {
                    v2bf pr;
                    pr[0] = (bf16_t)c0[j];
                    pr[1] = (bf16_t)c1[j];
                    *(v2bf*)&Bsh[(bN8 + j) * BSTR + bK] = pr;   // 4B aligned (bK even)
                }
            }
            __syncthreads();

            // ---------- compute: 2(M) x 4(N) WMMA tiles per wave ----------
            v16bf afrag[2];
            #pragma unroll
            for (int mt = 0; mt < 2; ++mt) {
                const bf16_t* ar = &Ash[(wm * 32 + mt * 16 + l16) * ASTR];
                // lanes 0-15: K 0..7 / 16..23 ; lanes 16-31: K 8..15 / 24..31
                v8bf a0 = *(const v8bf*)(ar + 8 * hlf);
                v8bf a1 = *(const v8bf*)(ar + 16 + 8 * hlf);
                afrag[mt] = __builtin_shufflevector(a0, a1,
                    0,1,2,3,4,5,6,7, 8,9,10,11,12,13,14,15);
            }
            #pragma unroll
            for (int nt = 0; nt < 4; ++nt) {
                const bf16_t* bc = &Bsh[(wn * 64 + nt * 16 + l16) * BSTR];
                // lanes 0-15: K 0..7 / 16..23 ; lanes 16-31: K 8..15 / 24..31
                v8bf b0 = *(const v8bf*)(bc + 8 * hlf);
                v8bf b1 = *(const v8bf*)(bc + 16 + 8 * hlf);
                v16bf bfrag = __builtin_shufflevector(b0, b1,
                    0,1,2,3,4,5,6,7, 8,9,10,11,12,13,14,15);
                acc[0][nt] = __builtin_amdgcn_wmma_f32_16x16x32_bf16(
                    false, afrag[0], false, bfrag, (short)0, acc[0][nt], false, false);
                acc[1][nt] = __builtin_amdgcn_wmma_f32_16x16x32_bf16(
                    false, afrag[1], false, bfrag, (short)0, acc[1][nt], false, false);
            }
            __syncthreads();
        }
    }

    // ---------- epilogue: + (w @ bias), write out ----------
    float bv[4][EXPERTS];
    #pragma unroll
    for (int nt = 0; nt < 4; ++nt) {
        const int gcol = blockN + wn * 64 + nt * 16 + l16;
        #pragma unroll
        for (int e2 = 0; e2 < EXPERTS; ++e2)
            bv[nt][e2] = bias[(size_t)e2 * OUT_DIM + gcol];
    }
    #pragma unroll
    for (int mt = 0; mt < 2; ++mt) {
        #pragma unroll
        for (int r = 0; r < 8; ++r) {
            // C/D layout: lanes 0-15 hold rows 0..7 (per VGPR), lanes 16-31 rows 8..15
            const int grow = blockM + wm * 32 + mt * 16 + hlf * 8 + r;
            const float4* wrp = (const float4*)(gate + (size_t)grow * EXPERTS);
            float4 w0 = wrp[0], w1 = wrp[1];
            float wv[8] = { w0.x, w0.y, w0.z, w0.w, w1.x, w1.y, w1.z, w1.w };
            #pragma unroll
            for (int nt = 0; nt < 4; ++nt) {
                float bsum = 0.f;
                #pragma unroll
                for (int e2 = 0; e2 < EXPERTS; ++e2) bsum += wv[e2] * bv[nt][e2];
                const int gcol = blockN + wn * 64 + nt * 16 + l16;
                out[(size_t)grow * OUT_DIM + gcol] = acc[mt][nt][r] + bsum;
            }
        }
    }
}

extern "C" void kernel_launch(void* const* d_in, const int* in_sizes, int n_in,
                              void* d_out, int out_size, void* d_ws, size_t ws_size,
                              hipStream_t stream) {
    (void)in_sizes; (void)n_in; (void)out_size; (void)d_ws; (void)ws_size;
    const float* x    = (const float*)d_in[0];   // [B, K]
    const float* gate = (const float*)d_in[1];   // [B, E]
    const float* W    = (const float*)d_in[2];   // [E, K, N]
    const float* bias = (const float*)d_in[3];   // [E, 1, N]
    float* out        = (float*)d_out;           // [B, N]

    dim3 grid(OUT_DIM / BN, BATCH / BM);   // (8, 128)
    moe_linear_bf16_wmma<<<grid, 256, 0, stream>>>(x, gate, W, bias, out);
}
